// Model_51384988729809
// MI455X (gfx1250) — compile-verified, matrix operand order checked by reading, not run on hardware
//
#include <hip/hip_runtime.h>
#include <hip/hip_bf16.h>

// ---------------------------------------------------------------------------
// EdgeConv / ParticleNet pipeline for MI455X (gfx1250, wave32, WMMA).
// Math-level optimizations (exact w.r.t. reference):
//  * theta(h_i - h_j) = (h_i@tW) - (h_j@tW) + tb  -> per-NODE GEMM + gather
//  * phi_en(en_j) depends only on node j          -> per-NODE MLP + gather
//  * layer-2 h (theta/phi/BN/max) never reaches the output -> skipped
// Heavy parts (edge energy MLP 64->128->32, KNN Gram, node GEMM) run on
// v_wmma_f32_16x16x32_f16; KNN candidate tiles are double-buffered into LDS
// with gfx1250 async-to-LDS loads (ASYNCcnt) when the toolchain exposes them.
// ---------------------------------------------------------------------------

#define NODES 16384
#define NB 8
#define NN 2048

typedef _Float16 v16h __attribute__((ext_vector_type(16)));
typedef _Float16 h8   __attribute__((ext_vector_type(8)));
typedef float    v8f  __attribute__((ext_vector_type(8)));
typedef int      v4i  __attribute__((ext_vector_type(4)));

#if defined(__HIP_DEVICE_COMPILE__) && \
    __has_builtin(__builtin_amdgcn_global_load_async_to_lds_b128) && \
    __has_builtin(__builtin_amdgcn_s_wait_asynccnt)
#define HAVE_ASYNC_LDS 1
#else
#define HAVE_ASYNC_LDS 0
#endif

#if HAVE_ASYNC_LDS
typedef __attribute__((address_space(3))) v4i lds_v4i;
typedef __attribute__((address_space(1))) v4i gbl_v4i;
__device__ __forceinline__ lds_v4i* to_lds(void* p) {
    // generic LDS pointers carry the LDS byte offset in the low 32 bits
    return (lds_v4i*)(unsigned int)(unsigned long long)p;
}
__device__ __forceinline__ gbl_v4i* to_gbl(const void* p) {
    return (gbl_v4i*)(unsigned long long)p;
}
#define WAIT_ASYNC(n) __builtin_amdgcn_s_wait_asynccnt(n)
#else
#define WAIT_ASYNC(n)
#endif

__device__ __forceinline__ float lrelu(float x, float s) { return x > 0.f ? x : x * s; }

__device__ __forceinline__ v8f wmma_f16(v16h a, v16h b, v8f c) {
    // (neg_a, A, neg_b, B, c_mod, C, reuse_a, reuse_b)
    return __builtin_amdgcn_wmma_f32_16x16x32_f16(false, a, false, b, (short)0, c, false, false);
}

// CDNA5 16-bit A/B fragment layout: lane L holds K = {kb..kb+7} U {kb+16..kb+23},
// kb = (L&16)?8:0 -> two contiguous 8-half (16B) runs from a row-major row.
__device__ __forceinline__ v16h load_frag_row(const _Float16* row, int lane) {
    int kb = (lane & 16) ? 8 : 0;
    h8 lo = *(const h8*)(row + kb);
    h8 hi = *(const h8*)(row + kb + 16);
    return __builtin_shufflevector(lo, hi, 0,1,2,3,4,5,6,7,8,9,10,11,12,13,14,15);
}

// Register-resident top-16 (sorted ascending). Fully unrolled -> stays in VGPRs.
__device__ __forceinline__ void topk_insert(float (&bd)[16], int (&bi)[16], float d, int id) {
    if (d >= bd[15]) return;
    bd[15] = d; bi[15] = id;
#pragma unroll
    for (int t = 15; t > 0; --t) {
        if (bd[t] < bd[t - 1]) {
            float td = bd[t]; bd[t] = bd[t - 1]; bd[t - 1] = td;
            int   ti = bi[t]; bi[t] = bi[t - 1]; bi[t - 1] = ti;
        }
    }
}

// ------------------------------ conversions --------------------------------

// W[K][Nd] f32 -> WT[Nd][K] f16 (transposed so B-fragments load contiguously)
__global__ void conv_wt_f16(const float* __restrict__ W, _Float16* __restrict__ WT,
                            int K, int Nd) {
    int t = blockIdx.x * blockDim.x + threadIdx.x;
    if (t >= K * Nd) return;
    int k = t / Nd, n = t % Nd;
    WT[n * K + k] = (_Float16)W[t];
}

__global__ void h_to_f16_norms(const float* __restrict__ h, _Float16* __restrict__ hf,
                               float* __restrict__ norms, int F) {
    int i = blockIdx.x * blockDim.x + threadIdx.x;
    if (i >= NODES) return;
    float s = 0.f;
    for (int c = 0; c < F; ++c) {
        float v = h[i * F + c];
        hf[i * F + c] = (_Float16)v;
        s += v * v;
    }
    norms[i] = s;
}

__global__ void zero_f32(float* __restrict__ p, int n) {
    int i = blockIdx.x * blockDim.x + threadIdx.x;
    if (i < n) p[i] = 0.f;
}

// ------------------------------ KNN ----------------------------------------

// Layer 0: D=3, self-loops kept, k=16. Scalar path (3 MACs/candidate).
__global__ void knn3_kernel(const float* __restrict__ x, int* __restrict__ idx) {
    int g = blockIdx.x * blockDim.x + threadIdx.x;
    if (g >= NODES) return;
    int b = g / NN;
    float qx = x[g * 3], qy = x[g * 3 + 1], qz = x[g * 3 + 2];
    float bd[16]; int bi[16];
#pragma unroll
    for (int t = 0; t < 16; ++t) { bd[t] = 3.0e38f; bi[t] = g; }
    int base = b * NN;
    for (int c = 0; c < NN; ++c) {
        int cid = base + c;
        float dx = qx - x[cid * 3], dy = qy - x[cid * 3 + 1], dz = qz - x[cid * 3 + 2];
        topk_insert(bd, bi, dx * dx + dy * dy + dz * dz, cid);
    }
#pragma unroll
    for (int t = 0; t < 16; ++t) idx[g * 16 + t] = bi[t];
}

// Stage one 16xF f16 candidate tile (contiguous rows) into LDS.
template <int F>
__device__ __forceinline__ void stage_chunk(_Float16* dst, const _Float16* src, int lane) {
    constexpr int OPS = (16 * F * 2) / (32 * 16);  // per-wave async instructions
#pragma unroll
    for (int o = 0; o < OPS; ++o) {
        int byteOff = (o * 32 + lane) * 16;
#if HAVE_ASYNC_LDS
        __builtin_amdgcn_global_load_async_to_lds_b128(
            to_gbl((const char*)src + byteOff), to_lds((char*)dst + byteOff), 0, 0);
#else
        *(h8*)((char*)dst + byteOff) = *(const h8*)((const char*)src + byteOff);
#endif
    }
}

// Layers 1/2: F in {64,128}, exclude self, k=15. One wave per 16 queries.
// Query A-fragments live in registers; candidate tiles are double-buffered in
// LDS via async loads; Gram tile via WMMA; top-16 maintained in VGPRs.
template <int F>
__global__ __launch_bounds__(32) void knn_wmma_kernel(
        const _Float16* __restrict__ hf, const float* __restrict__ norms,
        int* __restrict__ idx) {
    constexpr int KS = F / 32;
    constexpr int OPS = (16 * F * 2) / (32 * 16);
    constexpr int NCH = NN / 16;
    __shared__ __attribute__((aligned(16))) _Float16 sC[2][16 * F];
    __shared__ __attribute__((aligned(16))) float sG[16 * 16];
    __shared__ float sCN[16];
    int lane = threadIdx.x;
    int b = blockIdx.x / NCH;
    int tile = blockIdx.x % NCH;
    int qbase = b * NN + tile * 16;
    int m = lane & 15;
    int mrow = (lane & 16) ? 8 : 0;

    // query fragments: resident in VGPRs for the whole scan
    v16h afrag[KS];
#pragma unroll
    for (int ks = 0; ks < KS; ++ks)
        afrag[ks] = load_frag_row(hf + (size_t)(qbase + m) * F + ks * 32, lane);
    float qn = norms[qbase + m];

    float bd[16]; int bi[16];
#pragma unroll
    for (int t = 0; t < 16; ++t) { bd[t] = 3.0e38f; bi[t] = qbase; }

    stage_chunk<F>(sC[0], hf + (size_t)(b * NN) * F, lane);
    for (int chunk = 0; chunk < NCH; ++chunk) {
        int cur = chunk & 1;
        int cbase = b * NN + chunk * 16;
        if (chunk + 1 < NCH) {
            stage_chunk<F>(sC[cur ^ 1], hf + (size_t)(cbase + 16) * F, lane);
            WAIT_ASYNC(OPS);  // current tile complete (async loads finish in order)
        } else {
            WAIT_ASYNC(0);
        }
        __syncthreads();

        v8f acc = {};
#pragma unroll
        for (int ks = 0; ks < KS; ++ks) {
            v16h bb = load_frag_row(sC[cur] + m * F + ks * 32, lane);
            acc = wmma_f16(afrag[ks], bb, acc);
        }
        if (lane < 16) sCN[lane] = norms[cbase + lane];
#pragma unroll
        for (int v = 0; v < 8; ++v) sG[(v + mrow) * 16 + m] = acc[v];
        __syncthreads();
        if (lane < 16) {
            for (int n = 0; n < 16; ++n) {
                int cid = cbase + n;
                if (cid == qbase + lane) continue;  // remove_self_loop
                float d = qn + sCN[n] - 2.f * sG[lane * 16 + n];
                topk_insert(bd, bi, d, cid);
            }
        }
        __syncthreads();
    }
    if (lane < 16) {
#pragma unroll
        for (int t = 0; t < 16; ++t) idx[(qbase + lane) * 16 + t] = bi[t];
    }
}

// ------------------------- node projections (theta/phi) --------------------

// Layer 0, din=3: scalar.
__global__ void nodeproj3_kernel(const float* __restrict__ x,
                                 const float* __restrict__ tW, const float* __restrict__ pW,
                                 const float* __restrict__ pb,
                                 float* __restrict__ T, float* __restrict__ P, int dout) {
    int i = blockIdx.x, c = threadIdx.x;
    float x0 = x[i * 3], x1 = x[i * 3 + 1], x2 = x[i * 3 + 2];
    T[i * dout + c] = x0 * tW[c] + x1 * tW[dout + c] + x2 * tW[2 * dout + c];
    P[i * dout + c] = x0 * pW[c] + x1 * pW[dout + c] + x2 * pW[2 * dout + c] + pb[c];
}

// Layer 1: WMMA GEMM, T = h@tW (no bias; tb folded into edge kernel), P = h@pW + pb.
__global__ __launch_bounds__(32) void nodeproj_wmma_kernel(
        const _Float16* __restrict__ hf,
        const _Float16* __restrict__ tWT, const _Float16* __restrict__ pWT,
        const float* __restrict__ pb,
        float* __restrict__ T, float* __restrict__ P, int din, int dout) {
    int lane = threadIdx.x;
    int mt = blockIdx.x, nt = blockIdx.y;
    int m = lane & 15;
    int mrow = (lane & 16) ? 8 : 0;
    v8f at = {}, ap = {};
    for (int ks = 0; ks < din; ks += 32) {
        v16h a  = load_frag_row(hf  + (size_t)(mt * 16 + m) * din + ks, lane);
        v16h bt = load_frag_row(tWT + (size_t)(nt * 16 + m) * din + ks, lane);
        v16h bp = load_frag_row(pWT + (size_t)(nt * 16 + m) * din + ks, lane);
        at = wmma_f16(a, bt, at);
        ap = wmma_f16(a, bp, ap);
    }
#pragma unroll
    for (int v = 0; v < 8; ++v) {
        int M = mt * 16 + v + mrow;
        int Nc = nt * 16 + m;
        T[(size_t)M * dout + Nc] = at[v];
        P[(size_t)M * dout + Nc] = ap[v] + pb[Nc];
    }
}

// ----------------------- edge message + BatchNorm + max --------------------

// stats[0:dout]=sum, [dout:2dout]=sumsq, [2dout:3dout]=mu, [3dout:4dout]=rsig
// 8 nodes per block -> 8x fewer same-address float atomics.
__global__ void edge_stats_kernel(const float* __restrict__ T, const float* __restrict__ P,
                                  const float* __restrict__ tb, const int* __restrict__ idx,
                                  int kk, int dout, float* __restrict__ stats) {
    int c = threadIdx.x;
    int i0 = blockIdx.x * 8;
    float tbc = tb[c];
    float s = 0.f, s2 = 0.f;
    for (int r = 0; r < 8; ++r) {
        int i = i0 + r;
        float ti = T[(size_t)i * dout + c] + tbc;
        for (int mEdge = 0; mEdge < kk; ++mEdge) {
            int j = idx[i * 16 + mEdge];
            float e = ti - T[(size_t)j * dout + c] + P[(size_t)j * dout + c];
            s += e; s2 += e * e;
        }
    }
    unsafeAtomicAdd(&stats[c], s);
    unsafeAtomicAdd(&stats[dout + c], s2);
}

__global__ void bn_finalize_kernel(float* __restrict__ stats, int dout, float invcnt) {
    int c = threadIdx.x;
    if (c >= dout) return;
    float mu = stats[c] * invcnt;
    float var = stats[dout + c] * invcnt - mu * mu;
    stats[2 * dout + c] = mu;
    stats[3 * dout + c] = rsqrtf(var + 1e-5f);
}

__global__ void edge_apply_kernel(const float* __restrict__ T, const float* __restrict__ P,
                                  const float* __restrict__ tb, const int* __restrict__ idx,
                                  int kk, int dout, const float* __restrict__ stats,
                                  const float* __restrict__ gamma, const float* __restrict__ beta,
                                  float* __restrict__ hout) {
    int i = blockIdx.x, c = threadIdx.x;
    float ti = T[(size_t)i * dout + c] + tb[c];
    float mu = stats[2 * dout + c], rs = stats[3 * dout + c];
    float g = gamma[c], be = beta[c];
    float mv = -3.0e38f;
    for (int mEdge = 0; mEdge < kk; ++mEdge) {
        int j = idx[i * 16 + mEdge];
        float e = ti - T[(size_t)j * dout + c] + P[(size_t)j * dout + c];
        float v = g * (e - mu) * rs + be;
        mv = fmaxf(mv, v);
    }
    hout[(size_t)i * dout + c] = lrelu(mv, 0.2f);  // NEG_H
}

// --------------------------- per-edge energy MLP ---------------------------

// 16 scalars (one per lane-m) -> MLP 1->64->128->32->1 on WMMA, slope -0.5.
__device__ __forceinline__ float en_mlp_tile(
        int lane, float s,
        const float* __restrict__ W1, const float* __restrict__ b1,
        const _Float16* __restrict__ W2T, const float* __restrict__ b2,
        const _Float16* __restrict__ W3T, const float* __restrict__ b3,
        const float* __restrict__ W4, const float* __restrict__ b4,
        _Float16* sh1, _Float16* sh2, float* sh3) {
    const float SL = -0.5f;  // NEG_MSG
    int m = lane & 15;
    int half = lane >> 4;
    int mrow = (lane & 16) ? 8 : 0;

    // h1 = leaky(s*W1 + b1) : 16x64 tile; lanes (m,half) split the 64 columns
    for (int c2 = 0; c2 < 32; ++c2) {
        int c = half * 32 + c2;
        sh1[m * 64 + c] = (_Float16)lrelu(s * W1[c] + b1[c], SL);
    }
    __syncthreads();

    // 16x64 @ 64x128
    v8f acc[8] = {};
    for (int ks = 0; ks < 64; ks += 32) {
        v16h a = load_frag_row(sh1 + m * 64 + ks, lane);
#pragma unroll
        for (int t = 0; t < 8; ++t) {
            v16h bb = load_frag_row(W2T + (t * 16 + m) * 64 + ks, lane);
            acc[t] = wmma_f16(a, bb, acc[t]);
        }
    }
#pragma unroll
    for (int t = 0; t < 8; ++t)
#pragma unroll
        for (int v = 0; v < 8; ++v) {
            int M = v + mrow, Nc = t * 16 + m;
            sh2[M * 128 + Nc] = (_Float16)lrelu(acc[t][v] + b2[Nc], SL);
        }
    __syncthreads();

    // 16x128 @ 128x32
    v8f acc2[2] = {};
    for (int ks = 0; ks < 128; ks += 32) {
        v16h a = load_frag_row(sh2 + m * 128 + ks, lane);
#pragma unroll
        for (int t = 0; t < 2; ++t) {
            v16h bb = load_frag_row(W3T + (t * 16 + m) * 128 + ks, lane);
            acc2[t] = wmma_f16(a, bb, acc2[t]);
        }
    }
#pragma unroll
    for (int t = 0; t < 2; ++t)
#pragma unroll
        for (int v = 0; v < 8; ++v) {
            int M = v + mrow, Nc = t * 16 + m;
            sh3[M * 32 + Nc] = lrelu(acc2[t][v] + b3[Nc], SL);
        }
    __syncthreads();

    float out = b4[0];
    for (int c = 0; c < 32; ++c) out += sh3[m * 32 + c] * W4[c];
    return out;
}

// phi_en per NODE (factored out of the edge loop). One wave = 16 nodes.
__global__ __launch_bounds__(32) void node_phi_en_kernel(
        const float* __restrict__ en,
        const float* __restrict__ W1, const float* __restrict__ b1,
        const _Float16* __restrict__ W2T, const float* __restrict__ b2,
        const _Float16* __restrict__ W3T, const float* __restrict__ b3,
        const float* __restrict__ W4, const float* __restrict__ b4,
        float* __restrict__ outv) {
    __shared__ __attribute__((aligned(16))) _Float16 sh1[16 * 64];
    __shared__ __attribute__((aligned(16))) _Float16 sh2[16 * 128];
    __shared__ __attribute__((aligned(16))) float    sh3[16 * 32];
    int lane = threadIdx.x;
    int base = blockIdx.x * 16;
    int m = lane & 15;
    float s = en[base + m];
    float o = en_mlp_tile(lane, s, W1, b1, W2T, b2, W3T, b3, W4, b4, sh1, sh2, sh3);
    if (lane < 16) outv[base + m] = o;
}

// theta_en per EDGE + mean-reduce. One wave = one node's <=16 edges.
__global__ __launch_bounds__(32) void edge_energy_kernel(
        const float* __restrict__ en, const int* __restrict__ idx, int kk,
        const float* __restrict__ W1, const float* __restrict__ b1,
        const _Float16* __restrict__ W2T, const float* __restrict__ b2,
        const _Float16* __restrict__ W3T, const float* __restrict__ b3,
        const float* __restrict__ W4, const float* __restrict__ b4,
        const float* __restrict__ phiNode,
        float* __restrict__ en_next, float* __restrict__ hs, int layer) {
    __shared__ __attribute__((aligned(16))) _Float16 sh1[16 * 64];
    __shared__ __attribute__((aligned(16))) _Float16 sh2[16 * 128];
    __shared__ __attribute__((aligned(16))) float    sh3[16 * 32];
    __shared__ float sred[16];
    int lane = threadIdx.x;
    int i = blockIdx.x;
    int m = lane & 15;
    int j = idx[i * 16 + (m < kk ? m : 0)];
    float s = en[i] - en[j];
    float o = en_mlp_tile(lane, s, W1, b1, W2T, b2, W3T, b3, W4, b4, sh1, sh2, sh3);
    if (lane < 16) sred[m] = (m < kk) ? (o + phiNode[j]) : 0.f;
    __syncthreads();
    if (lane == 0) {
        float acc = 0.f;
        for (int t = 0; t < 16; ++t) acc += sred[t];
        acc /= (float)kk;
        en_next[i] = acc;
        hs[i * 3 + layer] = acc;
    }
}

// ------------------------------ head + scatter -----------------------------

__global__ void proj_kernel(const float* __restrict__ hs,
                            const float* W0, const float* B0, const float* W1, const float* B1,
                            const float* W2, const float* B2, const float* W3, const float* B3,
                            const float* W4, const float* B4, const float* W5, const float* B5,
                            float* __restrict__ outv) {
    int i = blockIdx.x * blockDim.x + threadIdx.x;
    if (i >= NODES) return;
    const float* Ws[6] = {W0, W1, W2, W3, W4, W5};
    const float* Bs[6] = {B0, B1, B2, B3, B4, B5};
    const int dims[7] = {3, 20, 40, 30, 10, 5, 1};
    float a[40], b[40];
    a[0] = hs[i * 3]; a[1] = hs[i * 3 + 1]; a[2] = hs[i * 3 + 2];
    float* cur = a; float* nxt = b;
    for (int l = 0; l < 6; ++l) {
        int di = dims[l], dn = dims[l + 1];
        const float* W = Ws[l]; const float* bb = Bs[l];
        for (int oc = 0; oc < dn; ++oc) {
            float acc = bb[oc];
            for (int k = 0; k < di; ++k) acc += cur[k] * W[k * dn + oc];
            nxt[oc] = (l < 5) ? lrelu(acc, -0.8f) : acc;  // NEG_PROJ
        }
        float* t = cur; cur = nxt; nxt = t;
    }
    outv[i] = cur[0];
}

__global__ void scatter_kernel(const float* __restrict__ e, const int* __restrict__ x_idx,
                               float* __restrict__ outp) {
    int g = blockIdx.x * blockDim.x + threadIdx.x;
    if (g >= NODES) return;
    int ch = x_idx[g * 3], i1 = x_idx[g * 3 + 1], i2 = x_idx[g * 3 + 2];
    if (ch >= 1 && ch < 7) {
        int b = g / NN;
        outp[((size_t)(b * 6 + (ch - 1))) * 4096 + i1 * 64 + i2] = e[g];
    }
}

// ------------------------------- driver ------------------------------------

extern "C" void kernel_launch(void* const* d_in, const int* in_sizes, int n_in,
                              void* d_out, int out_size, void* d_ws, size_t ws_size,
                              hipStream_t stream) {
    (void)in_sizes; (void)n_in; (void)out_size; (void)ws_size;
    const float* x    = (const float*)d_in[0];
    const float* x_en = (const float*)d_in[1];
    const int*   xidx = (const int*)d_in[2];
    // d_in[3] = length (always full N)

    // Param layout (setup_inputs insertion order), per layer l at base 4+22l:
    //  +0 thetaW +1 thetab +2 phiW +3 phib
    //  +4..11 theta_en (W1,b1,W2,b2,W3,b3,W4,b4)
    //  +12..19 phi_en  (W1,b1,W2,b2,W3,b3,W4,b4)
    //  +20 bn_gamma +21 bn_beta ; proj pairs from 70.
    auto LP = [&](int l, int off) { return (const float*)d_in[4 + 22 * l + off]; };
    auto PR = [&](int off) { return (const float*)d_in[70 + off]; };

    char* w = (char*)d_ws;
    size_t off = 0;
    auto carve = [&](size_t bytes) -> char* {
        char* p = w + off;
        off += (bytes + 255) & ~(size_t)255;
        return p;
    };
    float*     hA       = (float*)carve((size_t)NODES * 128 * 4);
    float*     hB       = (float*)carve((size_t)NODES * 128 * 4);
    _Float16*  hf16     = (_Float16*)carve((size_t)NODES * 128 * 2);
    float*     norms    = (float*)carve(NODES * 4);
    float*     T        = (float*)carve((size_t)NODES * 128 * 4);
    float*     P        = (float*)carve((size_t)NODES * 128 * 4);
    int*       idxb     = (int*)carve((size_t)NODES * 16 * 4);
    float*     enA      = (float*)carve(NODES * 4);
    float*     enB      = (float*)carve(NODES * 4);
    float*     hs       = (float*)carve((size_t)NODES * 3 * 4);
    float*     phiNode  = (float*)carve(NODES * 4);
    float*     stats    = (float*)carve(4 * 256 * 4);
    _Float16*  tWT16    = (_Float16*)carve(128 * 128 * 2);
    _Float16*  pWT16    = (_Float16*)carve(128 * 128 * 2);
    _Float16*  thW2T    = (_Float16*)carve(128 * 64 * 2);
    _Float16*  thW3T    = (_Float16*)carve(32 * 128 * 2);
    _Float16*  phW2T    = (_Float16*)carve(128 * 64 * 2);
    _Float16*  phW3T    = (_Float16*)carve(32 * 128 * 2);
    float*     projOut  = (float*)carve(NODES * 4);

    auto conv_en = [&](int l) {
        conv_wt_f16<<<(64 * 128 + 255) / 256, 256, 0, stream>>>(LP(l, 6), thW2T, 64, 128);
        conv_wt_f16<<<(128 * 32 + 255) / 256, 256, 0, stream>>>(LP(l, 8), thW3T, 128, 32);
        conv_wt_f16<<<(64 * 128 + 255) / 256, 256, 0, stream>>>(LP(l, 14), phW2T, 64, 128);
        conv_wt_f16<<<(128 * 32 + 255) / 256, 256, 0, stream>>>(LP(l, 16), phW3T, 128, 32);
    };
    auto energy_layer = [&](int l, const float* enCur, int kk, float* enNext) {
        conv_en(l);
        node_phi_en_kernel<<<NODES / 16, 32, 0, stream>>>(
            enCur, LP(l, 12), LP(l, 13), phW2T, LP(l, 15), phW3T, LP(l, 17), LP(l, 18), LP(l, 19),
            phiNode);
        edge_energy_kernel<<<NODES, 32, 0, stream>>>(
            enCur, idxb, kk, LP(l, 4), LP(l, 5), thW2T, LP(l, 7), thW3T, LP(l, 9), LP(l, 10),
            LP(l, 11), phiNode, enNext, hs, l);
    };

    // ---------------- Layer 0: din=3, dout=64, k=16 (self-loops kept) ------
    knn3_kernel<<<NODES / 256, 256, 0, stream>>>(x, idxb);
    nodeproj3_kernel<<<NODES, 64, 0, stream>>>(x, LP(0, 0), LP(0, 2), LP(0, 3), T, P, 64);
    zero_f32<<<1, 128, 0, stream>>>(stats, 128);
    edge_stats_kernel<<<NODES / 8, 64, 0, stream>>>(T, P, LP(0, 1), idxb, 16, 64, stats);
    bn_finalize_kernel<<<1, 64, 0, stream>>>(stats, 64, 1.f / (NB * NN * 16.f));
    edge_apply_kernel<<<NODES, 64, 0, stream>>>(T, P, LP(0, 1), idxb, 16, 64, stats,
                                                LP(0, 20), LP(0, 21), hA);
    energy_layer(0, x_en, 16, enA);

    // ---------------- Layer 1: din=64, dout=128, k=15 ----------------------
    h_to_f16_norms<<<NODES / 256, 256, 0, stream>>>(hA, hf16, norms, 64);
    knn_wmma_kernel<64><<<NB * (NN / 16), 32, 0, stream>>>(hf16, norms, idxb);
    conv_wt_f16<<<(64 * 128 + 255) / 256, 256, 0, stream>>>(LP(1, 0), tWT16, 64, 128);
    conv_wt_f16<<<(64 * 128 + 255) / 256, 256, 0, stream>>>(LP(1, 2), pWT16, 64, 128);
    nodeproj_wmma_kernel<<<dim3(NODES / 16, 128 / 16), 32, 0, stream>>>(
        hf16, tWT16, pWT16, LP(1, 3), T, P, 64, 128);
    zero_f32<<<1, 256, 0, stream>>>(stats, 256);
    edge_stats_kernel<<<NODES / 8, 128, 0, stream>>>(T, P, LP(1, 1), idxb, 15, 128, stats);
    bn_finalize_kernel<<<1, 128, 0, stream>>>(stats, 128, 1.f / (NB * NN * 15.f));
    edge_apply_kernel<<<NODES, 128, 0, stream>>>(T, P, LP(1, 1), idxb, 15, 128, stats,
                                                 LP(1, 20), LP(1, 21), hB);
    energy_layer(1, enA, 15, enB);

    // ---------------- Layer 2: only the energy path reaches the output -----
    h_to_f16_norms<<<NODES / 256, 256, 0, stream>>>(hB, hf16, norms, 128);
    knn_wmma_kernel<128><<<NB * (NN / 16), 32, 0, stream>>>(hf16, norms, idxb);
    energy_layer(2, enB, 15, enA);

    // ---------------- Projection head + scatter ----------------------------
    proj_kernel<<<NODES / 256, 256, 0, stream>>>(
        hs, PR(0), PR(1), PR(2), PR(3), PR(4), PR(5), PR(6), PR(7), PR(8), PR(9), PR(10), PR(11),
        projOut);
    zero_f32<<<(NB * 6 * 64 * 64 + 255) / 256, 256, 0, stream>>>((float*)d_out, NB * 6 * 64 * 64);
    scatter_kernel<<<NODES / 256, 256, 0, stream>>>(projOut, xidx, (float*)d_out);
}
